// SMPL_80058190398137
// MI455X (gfx1250) — matrix operand order repaired and checked by reading, not compiled, not run on hardware
//
#include <hip/hip_runtime.h>
#include <hip/hip_bf16.h>

typedef __attribute__((ext_vector_type(16))) _Float16 v16h;
typedef __attribute__((ext_vector_type(8)))  float    v8f;

#define V_NUM   100000
#define N_J     24
#define N_BETA  10
#define N_P     207

// workspace layout (float offsets)
#define OFF_R    0      // 24*9 = 216 floats: rotation matrices
#define OFF_LROT 256    // 207 floats: lrotmin
#define OFF_A    512    // 24*16 = 384 floats: skinning matrices A (flattened 4x4)
#define OFF_JP   1024   // JRED_BLOCKS*72 floats: per-block joint partials
#define JRED_BLOCKS 120

// ---------------- Kernel 1: Rodrigues + lrotmin (1 wave) ----------------
__global__ void k_rodrigues(const float* __restrict__ pose, float* __restrict__ ws) {
    const int j = threadIdx.x;
    if (j >= N_J) return;
    const float rx = pose[j*3+0], ry = pose[j*3+1], rz = pose[j*3+2];
    const float th  = sqrtf(rx*rx + ry*ry + rz*rz) + 1e-8f;
    const float inv = 1.0f / th;
    const float x = rx*inv, y = ry*inv, z = rz*inv;
    const float c = cosf(th), s = sinf(th), ic = 1.0f - c;
    float R[9];
    R[0] = c + ic*x*x;     R[1] = ic*x*y - s*z;  R[2] = ic*x*z + s*y;
    R[3] = ic*x*y + s*z;   R[4] = c + ic*y*y;    R[5] = ic*y*z - s*x;
    R[6] = ic*x*z - s*y;   R[7] = ic*y*z + s*x;  R[8] = c + ic*z*z;
    #pragma unroll
    for (int e = 0; e < 9; ++e) ws[OFF_R + j*9 + e] = R[e];
    if (j >= 1) {
        #pragma unroll
        for (int e = 0; e < 9; ++e)
            ws[OFF_LROT + (j-1)*9 + e] = R[e] - ((e==0 || e==4 || e==8) ? 1.0f : 0.0f);
    }
}

// ------- Kernel 2: v_shaped (on the fly) + J_regressor partial reduction -------
__global__ __launch_bounds__(256) void k_jreduce(
    const float* __restrict__ sdirs, const float* __restrict__ vtmpl,
    const float* __restrict__ jreg,  const float* __restrict__ beta,
    float* __restrict__ ws)
{
    float bt[N_BETA];
    #pragma unroll
    for (int b = 0; b < N_BETA; ++b) bt[b] = beta[b];

    float acc[N_J][3];
    #pragma unroll
    for (int j = 0; j < N_J; ++j) { acc[j][0]=0.f; acc[j][1]=0.f; acc[j][2]=0.f; }

    const int tid    = blockIdx.x * blockDim.x + threadIdx.x;
    const int stride = gridDim.x * blockDim.x;
    for (int v = tid; v < V_NUM; v += stride) {
        float vs[3];
        #pragma unroll
        for (int d = 0; d < 3; ++d) {
            const float* row = sdirs + (size_t)(v*3 + d) * N_BETA;
            float s = vtmpl[v*3 + d];
            #pragma unroll
            for (int b = 0; b < N_BETA; ++b) s += __builtin_nontemporal_load(row + b) * bt[b];
            vs[d] = s;
        }
        #pragma unroll
        for (int j = 0; j < N_J; ++j) {
            const float w = __builtin_nontemporal_load(jreg + (size_t)j * V_NUM + v);
            acc[j][0] += w * vs[0]; acc[j][1] += w * vs[1]; acc[j][2] += w * vs[2];
        }
    }

    // deterministic reduction: wave shuffle tree, then fixed-order LDS sum
    __shared__ float wred[8][72];
    const int lane = threadIdx.x & 31;
    const int wid  = threadIdx.x >> 5;
    #pragma unroll
    for (int j = 0; j < N_J; ++j)
        #pragma unroll
        for (int d = 0; d < 3; ++d) {
            float v = acc[j][d];
            #pragma unroll
            for (int m = 16; m >= 1; m >>= 1) v += __shfl_xor(v, m, 32);
            if (lane == 0) wred[wid][j*3 + d] = v;
        }
    __syncthreads();
    if (threadIdx.x < 72) {
        float s = 0.f;
        #pragma unroll
        for (int w = 0; w < 8; ++w) s += wred[w][threadIdx.x];
        ws[OFF_JP + blockIdx.x*72 + threadIdx.x] = s;
    }
}

// ---------------- Kernel 3: finish Jts + forward kinematics -> A ----------------
__global__ void k_fk(float* __restrict__ ws) {
    __shared__ float Jls[72];
    const int t = threadIdx.x;
    if (t < 72) {
        float s = 0.f;
        for (int b = 0; b < JRED_BLOCKS; ++b) s += ws[OFF_JP + b*72 + t];
        Jls[t] = s;
    }
    __syncthreads();
    if (t == 0) {
        const int PAR[N_J] = {-1,0,0,0,1,2,3,4,5,6,7,8,9,9,9,12,13,14,16,17,18,19,20,21};
        float G[N_J][16];
        for (int a = 0; a < 3; ++a) {
            for (int b = 0; b < 3; ++b) G[0][a*4+b] = ws[OFF_R + a*3 + b];
            G[0][a*4+3] = Jls[a];
        }
        G[0][12]=0.f; G[0][13]=0.f; G[0][14]=0.f; G[0][15]=1.f;
        for (int i = 1; i < N_J; ++i) {
            const int p = PAR[i];
            float loc[16];
            for (int a = 0; a < 3; ++a) {
                for (int b = 0; b < 3; ++b) loc[a*4+b] = ws[OFF_R + i*9 + a*3 + b];
                loc[a*4+3] = Jls[i*3+a] - Jls[p*3+a];
            }
            loc[12]=0.f; loc[13]=0.f; loc[14]=0.f; loc[15]=1.f;
            for (int a = 0; a < 4; ++a)
                for (int b = 0; b < 4; ++b) {
                    float s = 0.f;
                    for (int k = 0; k < 4; ++k) s += G[p][a*4+k] * loc[k*4+b];
                    G[i][a*4+b] = s;
                }
        }
        for (int j = 0; j < N_J; ++j)
            for (int a = 0; a < 4; ++a) {
                const float corr = G[j][a*4+0]*Jls[j*3+0] + G[j][a*4+1]*Jls[j*3+1]
                                 + G[j][a*4+2]*Jls[j*3+2];
                for (int b = 0; b < 4; ++b)
                    ws[OFF_A + j*16 + a*4 + b] = G[j][a*4+b] - ((b==3) ? corr : 0.f);
            }
    }
}

// ------- Kernel 4: posedirs NT stream + WMMA skinning (1 tile of 16 verts / wave) -------
__global__ __launch_bounds__(256) void k_vertex(
    const float* __restrict__ posedirs, const float* __restrict__ shapedirs,
    const float* __restrict__ v_template, const float* __restrict__ weights,
    const float* __restrict__ beta, const float* __restrict__ trans,
    const float* __restrict__ ws, float* __restrict__ out)
{
    __shared__ float wbuf[8][48 + 256];   // per-wave: v_posed[48] + T tile[16x16]

    const int lane = threadIdx.x & 31;
    const int wid  = threadIdx.x >> 5;
    const int tile = blockIdx.x * 8 + wid;
    if (tile >= V_NUM / 16) return;          // wave-uniform: EXEC stays all-ones for WMMA
    const int vb = tile * 16;

    float* vpose = wbuf[wid];
    float* Tls   = wbuf[wid] + 48;

    // lrotmin: each lane needs exactly lrot[lane + 32*i], i=0..6 -> keep in registers
    float lr[7];
    #pragma unroll
    for (int i = 0; i < 6; ++i) lr[i] = ws[OFF_LROT + lane + 32*i];
    lr[6] = (lane + 192 < N_P) ? ws[OFF_LROT + lane + 192] : 0.f;

    // v_shaped for the 48 (vertex,dim) rows of this tile
    float bt[N_BETA];
    #pragma unroll
    for (int b = 0; b < N_BETA; ++b) bt[b] = beta[b];
    for (int r = lane; r < 48; r += 32) {
        const int row = vb*3 + r;
        const float* srow = shapedirs + (size_t)row * N_BETA;
        float s = v_template[row];
        #pragma unroll
        for (int b = 0; b < N_BETA; ++b) s += srow[b] * bt[b];
        vpose[r] = s;
    }
    asm volatile("" ::: "memory");

    // dominant stream: 48 rows x 207 of posedirs; coalesced non-temporal b32 loads,
    // lrotmin entirely in registers (zero DS traffic in the hot loop)
    for (int r = 0; r < 48; ++r) {
        const float* prow = posedirs + (size_t)(vb*3 + r) * N_P;
        float acc = 0.f;
        #pragma unroll
        for (int i = 0; i < 6; ++i)
            acc += __builtin_nontemporal_load(prow + lane + 32*i) * lr[i];
        if (lane + 192 < N_P)
            acc += __builtin_nontemporal_load(prow + lane + 192) * lr[6];
        #pragma unroll
        for (int m = 16; m >= 1; m >>= 1) acc += __shfl_xor(acc, m, 32);
        if (lane == 0) vpose[r] += acc;
    }
    asm volatile("" ::: "memory");

    // LBS blend T = W(16x24) x A(24x16) via v_wmma_f32_16x16x32_f16,
    // fp32-accurate using hi/lo f16 operand splitting (4 chained WMMAs).
    const int nn = lane & 15;   // A: row M (vertex), B: col N, D: col N
    const int hh = lane >> 4;
    v16h aH, aL, bH, bL;
    #pragma unroll
    for (int e = 0; e < 16; ++e) {
        // A fragment (16x32 f16): lanes 0-15 hold K=0..7,16..23; lanes 16-31 K=8..15,24..31
        const int ka = e + 8*hh + ((e >= 8) ? 8 : 0);
        const float wv = (ka < N_J)
            ? __builtin_nontemporal_load(weights + (size_t)(vb + nn) * N_J + ka) : 0.f;
        const _Float16 whi = (_Float16)wv;
        aH[e] = whi;
        aL[e] = (_Float16)(wv - (float)whi);
        // B fragment (32x16 f16): lanes 0-15 hold K=0..7,16..23; lanes 16-31 K=8..15,24..31
        const int kb = e + 8*hh + ((e >= 8) ? 8 : 0);
        const float av = (kb < N_J) ? ws[OFF_A + kb*16 + nn] : 0.f;
        const _Float16 ahi = (_Float16)av;
        bH[e] = ahi;
        bL[e] = (_Float16)(av - (float)ahi);
    }
    v8f dacc = {};
    dacc = __builtin_amdgcn_wmma_f32_16x16x32_f16(false, aL, false, bL, (short)0, dacc, false, false);
    dacc = __builtin_amdgcn_wmma_f32_16x16x32_f16(false, aH, false, bL, (short)0, dacc, false, false);
    dacc = __builtin_amdgcn_wmma_f32_16x16x32_f16(false, aL, false, bH, (short)0, dacc, false, false);
    dacc = __builtin_amdgcn_wmma_f32_16x16x32_f16(false, aH, false, bH, (short)0, dacc, false, false);

    // D layout: VGPR r holds row M = r + 8*half, col N = lane%16
    #pragma unroll
    for (int r = 0; r < 8; ++r) Tls[(r + 8*hh)*16 + nn] = dacc[r];
    asm volatile("" ::: "memory");

    // v = T * [v_posed, 1] + trans
    if (lane < 16) {
        const int m = lane;
        const float r0 = vpose[m*3+0], r1 = vpose[m*3+1], r2 = vpose[m*3+2];
        const float t0 = trans[0], t1 = trans[1], t2 = trans[2];
        #pragma unroll
        for (int a = 0; a < 3; ++a) {
            const float* Tr = Tls + m*16 + a*4;
            const float tv = (a == 0) ? t0 : (a == 1) ? t1 : t2;
            const float res = Tr[0]*r0 + Tr[1]*r1 + Tr[2]*r2 + Tr[3] + tv;
            __builtin_nontemporal_store(res, out + (size_t)(vb + m)*3 + a);
        }
    }
}

// ---------------- Host launcher ----------------
extern "C" void kernel_launch(void* const* d_in, const int* in_sizes, int n_in,
                              void* d_out, int out_size, void* d_ws, size_t ws_size,
                              hipStream_t stream) {
    const float* beta  = (const float*)d_in[0];
    const float* pose  = (const float*)d_in[1];
    const float* trans = (const float*)d_in[2];
    const float* sdirs = (const float*)d_in[3];
    const float* vtmpl = (const float*)d_in[4];
    const float* jreg  = (const float*)d_in[5];
    const float* pdirs = (const float*)d_in[6];
    const float* wts   = (const float*)d_in[7];
    const int*   faces = (const int*)d_in[8];
    float* out = (float*)d_out;
    float* ws  = (float*)d_ws;

    k_rodrigues<<<1, 32, 0, stream>>>(pose, ws);
    k_jreduce<<<JRED_BLOCKS, 256, 0, stream>>>(sdirs, vtmpl, jreg, beta, ws);
    k_fk<<<1, 128, 0, stream>>>(ws);

    const int tiles  = V_NUM / 16;         // 6250 tiles of 16 vertices
    const int blocks = (tiles + 7) / 8;    // 8 waves (tiles) per 256-thread block
    k_vertex<<<blocks, 256, 0, stream>>>(pdirs, sdirs, vtmpl, wts, beta, trans, ws, out);

    // faces passthrough into the output tail (int bits preserved)
    hipMemcpyAsync(out + (size_t)V_NUM*3, faces, (size_t)in_sizes[8]*sizeof(int),
                   hipMemcpyDeviceToDevice, stream);
}
